// LowRankRNN_19275813225191
// MI455X (gfx1250) — matrix operand order repaired and checked by reading, not compiled, not run on hardware
//
#include <hip/hip_runtime.h>

// Problem constants (match reference)
#define Bv 256
#define Tv 1024
#define Iv 64
#define Hv 512
#define Ov 32
#define Rv 2
#define MTOT (Bv * Tv)   // 262144 rows

typedef __attribute__((ext_vector_type(16))) __bf16 v16bf;
typedef __attribute__((ext_vector_type(8)))  float  v8f;

// ---------------------------------------------------------------------------
// Kernel 1: xin = x @ Wi^T + bi    (written into the hidden region of d_out)
// GEMM M=B*T, N=H, K=I.  bf16 WMMA 16x16x32, f32 accumulate.
// Fragment layouts used here (A and B share a contiguous k order, not the ISA table order):
//   A (16x32, 16-bit): lane l -> row M=l&15; halves e=0..15 -> K = (l>=16?16:0)+e
//   B (32x16, 16-bit): lane l -> col N=l&15; halves e=0..15 -> K = (l>=16?16:0)+e
//   C (16x16, f32): vgpr r, lane l -> M = r + 8*(l>=16), N = l&15
// ---------------------------------------------------------------------------
__global__ __launch_bounds__(256)
void lrrnn_xin_gemm(const float* __restrict__ x, const float* __restrict__ Wi,
                    const float* __restrict__ bi, float* __restrict__ hidden) {
    const int m0   = blockIdx.x * 16;
    const int wave = threadIdx.x >> 5;
    const int lane = threadIdx.x & 31;
    const int mrow = m0 + (lane & 15);
    const int hi   = lane >> 4;          // 0 or 1 (half-wave)

    // A fragments for k0 = 0, 32  (I = 64)
    v16bf a[2];
    const float* xr = x + (size_t)mrow * Iv;
#pragma unroll
    for (int ks = 0; ks < 2; ++ks) {
        const float* p = xr + ks * 32 + hi * 16;
#pragma unroll
        for (int e = 0; e < 8; ++e) a[ks][e]     = (__bf16)p[e];
#pragma unroll
        for (int e = 0; e < 8; ++e) a[ks][8 + e] = (__bf16)p[8 + e];
    }

    const int n0w = wave * 64;           // 8 waves cover H = 512
#pragma unroll
    for (int nt = 0; nt < 4; ++nt) {
        const int n0   = n0w + nt * 16;
        const int ncol = n0 + (lane & 15);
        v8f c = {};
#pragma unroll
        for (int ks = 0; ks < 2; ++ks) {
            v16bf b;
            const float* wp = Wi + (size_t)ncol * Iv + ks * 32 + hi * 16;
#pragma unroll
            for (int e = 0; e < 16; ++e) b[e] = (__bf16)wp[e];
            c = __builtin_amdgcn_wmma_f32_16x16x32_bf16(
                    false, a[ks], false, b, (short)0, c, false, false);
        }
        const float bv = bi[ncol];
        float* op = hidden + (size_t)(m0 + hi * 8) * Hv + ncol;
#pragma unroll
        for (int r = 0; r < 8; ++r) op[(size_t)r * Hv] = c[r] + bv;
    }
}

// ---------------------------------------------------------------------------
// Kernel 2: sequential low-rank scan, in place over hidden (fp32 throughout).
// One wave32 per batch, 16 h-elements per lane (h index = lane*16 + e).
//  - rank-2 projection p = h@n reduced with a pure in-wave xor butterfly:
//    no barriers, no LDS, and every lane ends with the full sum.
//  - xin stream double-buffered in registers: the t+1 global_load_b128s are
//    issued before the reduction chain, so HBM latency is off the serial path.
// ---------------------------------------------------------------------------
__global__ __launch_bounds__(32)
void lrrnn_scan(const float* __restrict__ mfac, const float* __restrict__ nfac,
                float* __restrict__ hidden) {
    const int b    = blockIdx.x;
    const int lane = threadIdx.x;                     // 0..31
    float* hb = hidden + (size_t)b * Tv * Hv + lane * 16;

    float m0[16], m1[16], n0[16], n1[16], h[16], xc[16], xn[16];
#pragma unroll
    for (int e = 0; e < 16; ++e) {
        const int idx = lane * 16 + e;
        m0[e] = mfac[idx * Rv + 0];
        m1[e] = mfac[idx * Rv + 1];
        n0[e] = nfac[idx * Rv + 0];
        n1[e] = nfac[idx * Rv + 1];
        h[e]  = 0.0f;
    }

    // preload xin(t=0)
    {
        const float4* p = (const float4*)hb;
#pragma unroll
        for (int j = 0; j < 4; ++j) {
            float4 v = p[j];
            xc[4*j+0] = v.x; xc[4*j+1] = v.y; xc[4*j+2] = v.z; xc[4*j+3] = v.w;
        }
    }

    for (int t = 0; t < Tv; ++t) {
        // prefetch xin(t+1) -- independent of the recurrence chain
        if (t + 1 < Tv) {
            const float4* p = (const float4*)(hb + (size_t)(t + 1) * Hv);
#pragma unroll
            for (int j = 0; j < 4; ++j) {
                float4 v = p[j];
                xn[4*j+0] = v.x; xn[4*j+1] = v.y; xn[4*j+2] = v.z; xn[4*j+3] = v.w;
            }
        }

        // p_r = sum_h h * n[:,r]  (local 16-FMA dot + wave32 xor butterfly)
        float p0 = 0.0f, p1 = 0.0f;
#pragma unroll
        for (int e = 0; e < 16; ++e) {
            p0 = fmaf(h[e], n0[e], p0);
            p1 = fmaf(h[e], n1[e], p1);
        }
#pragma unroll
        for (int off = 16; off > 0; off >>= 1) {
            p0 += __shfl_xor(p0, off, 32);
            p1 += __shfl_xor(p1, off, 32);
        }
        const float s0 = p0 * (1.0f / (float)Hv);
        const float s1 = p1 * (1.0f / (float)Hv);

        // h(t) = relu(xin(t) + s0*m0 + s1*m1); write back in place
#pragma unroll
        for (int e = 0; e < 16; ++e)
            h[e] = fmaxf(fmaf(s0, m0[e], fmaf(s1, m1[e], xc[e])), 0.0f);

        float4* op = (float4*)(hb + (size_t)t * Hv);
#pragma unroll
        for (int j = 0; j < 4; ++j) {
            float4 v;
            v.x = h[4*j+0]; v.y = h[4*j+1]; v.z = h[4*j+2]; v.w = h[4*j+3];
            op[j] = v;
        }

#pragma unroll
        for (int e = 0; e < 16; ++e) xc[e] = xn[e];
    }
}

// ---------------------------------------------------------------------------
// Kernel 3: y = hidden @ Wo^T + bo   (M=B*T, N=32, K=512), bf16 WMMA.
// 8 waves per block; each wave owns a 16(M) x 32(N) tile, 16 K-steps.
// ---------------------------------------------------------------------------
__global__ __launch_bounds__(256)
void lrrnn_out_gemm(const float* __restrict__ hidden, const float* __restrict__ Wo,
                    const float* __restrict__ bo, float* __restrict__ y) {
    const int wave = threadIdx.x >> 5;
    const int lane = threadIdx.x & 31;
    const int m0   = blockIdx.x * 128 + wave * 16;
    const int hi   = lane >> 4;
    const int mrow = m0 + (lane & 15);
    const int ncol = lane & 15;

    const float* hr = hidden + (size_t)mrow * Hv;
    v8f c0 = {}, c1 = {};

    for (int ks = 0; ks < 16; ++ks) {
        v16bf a;
        const float* p = hr + ks * 32 + hi * 16;
#pragma unroll
        for (int e = 0; e < 8; ++e) a[e]     = (__bf16)p[e];
#pragma unroll
        for (int e = 0; e < 8; ++e) a[8 + e] = (__bf16)p[8 + e];

        v16bf b0, b1;
        const float* w0 = Wo + (size_t)ncol        * Hv + ks * 32 + hi * 16;
        const float* w1 = Wo + (size_t)(16 + ncol) * Hv + ks * 32 + hi * 16;
#pragma unroll
        for (int e = 0; e < 16; ++e) b0[e] = (__bf16)w0[e];
#pragma unroll
        for (int e = 0; e < 16; ++e) b1[e] = (__bf16)w1[e];

        c0 = __builtin_amdgcn_wmma_f32_16x16x32_bf16(
                 false, a, false, b0, (short)0, c0, false, false);
        c1 = __builtin_amdgcn_wmma_f32_16x16x32_bf16(
                 false, a, false, b1, (short)0, c1, false, false);
    }

    const float bv0 = bo[ncol];
    const float bv1 = bo[16 + ncol];
    float* yp = y + (size_t)(m0 + hi * 8) * Ov;
#pragma unroll
    for (int r = 0; r < 8; ++r) {
        yp[(size_t)r * Ov + ncol]      = c0[r] + bv0;
        yp[(size_t)r * Ov + 16 + ncol] = c1[r] + bv1;
    }
}

// ---------------------------------------------------------------------------
extern "C" void kernel_launch(void* const* d_in, const int* in_sizes, int n_in,
                              void* d_out, int out_size, void* d_ws, size_t ws_size,
                              hipStream_t stream) {
    const float* x  = (const float*)d_in[0];  // [B,T,I]
    const float* Wi = (const float*)d_in[1];  // [H,I]
    const float* bi = (const float*)d_in[2];  // [H]
    const float* m  = (const float*)d_in[3];  // [H,R]
    const float* n  = (const float*)d_in[4];  // [H,R]
    const float* Wo = (const float*)d_in[5];  // [O,H]
    const float* bo = (const float*)d_in[6];  // [O]

    float* y      = (float*)d_out;                       // [B,T,O]
    float* hidden = y + (size_t)Bv * Tv * Ov;            // [B,T,H]

    // Phase 1: xin -> hidden region (bf16 WMMA GEMM)
    lrrnn_xin_gemm<<<MTOT / 16, 256, 0, stream>>>(x, Wi, bi, hidden);
    // Phase 2: sequential low-rank recurrence, one wave32 per batch, in place
    lrrnn_scan<<<Bv, 32, 0, stream>>>(m, n, hidden);
    // Phase 3: output projection (bf16 WMMA GEMM)
    lrrnn_out_gemm<<<MTOT / 128, 256, 0, stream>>>(hidden, Wo, bo, y);
}
